// CWRNNLM_10531259810486
// MI455X (gfx1250) — matrix operand order repaired and checked by reading, not compiled, not run on hardware
//
#include <hip/hip_runtime.h>
#include <hip/hip_bf16.h>

typedef __attribute__((ext_vector_type(16))) __bf16 v16bf;
typedef __attribute__((ext_vector_type(8)))  __bf16 v8bf;
typedef __attribute__((ext_vector_type(8)))  float  v8f;

#define Bsz 16
#define Tlen 255
#define Edim 256
#define NH 256
#define Vocab 32000

// Combine two contiguous 8x bf16 (16B) loads into one WMMA 16-element operand.
__device__ inline v16bf ld_frag(const __bf16* p0, const __bf16* p1) {
    union { v16bf v; v8bf h[2]; } u;
    u.h[0] = *(const v8bf*)p0;
    u.h[1] = *(const v8bf*)p1;
    return u.v;
}

// ---------------------------------------------------------------------------
// Kernel W: Wo (256 x 32000 f32, row-major) -> WoT (32000 x 256 bf16).
// K becomes contiguous so B-fragments are two b128 loads per lane.
// ---------------------------------------------------------------------------
__global__ void wo_transpose_kernel(const float* __restrict__ Wo,
                                    __bf16* __restrict__ WoT) {
    __shared__ float tile[32][33];
    const int n0 = blockIdx.x * 32;          // along V
    const int k0 = blockIdx.y * 32;          // along NH
    const int tx = threadIdx.x;              // 0..31
    const int ty = threadIdx.y;              // 0..7
#pragma unroll
    for (int r = 0; r < 4; ++r) {
        int k = ty + r * 8;
        tile[k][tx] = Wo[(size_t)(k0 + k) * Vocab + n0 + tx];
    }
    __syncthreads();
#pragma unroll
    for (int r = 0; r < 4; ++r) {
        int n = ty + r * 8;
        WoT[(size_t)(n0 + n) * NH + k0 + tx] = (__bf16)tile[tx][n];
    }
}

// ---------------------------------------------------------------------------
// Kernel A: clockwork RNN recurrence. ONE workgroup (8 waves) keeps WiT+WhT
// resident in 320KB LDS as bf16 and runs all 255 sequential steps with
// v_wmma_f32_16x16x32_bf16 (M=16 == batch). Emits h_t as bf16 to workspace.
// Dynamic LDS layout (bytes):
//   sWiT 131072 | sWhT 131072 | sE 8192 | sH 8192 | sHf 16384 | sBias 1024
// ---------------------------------------------------------------------------
__global__ void cwrnn_recurrence_kernel(const int*   __restrict__ x,
                                        const float* __restrict__ emb,
                                        const float* __restrict__ Wi,
                                        const float* __restrict__ Wh,
                                        const float* __restrict__ bi,
                                        const float* __restrict__ bh,
                                        __bf16*      __restrict__ hbf_out) {
    extern __shared__ char smem[];
    __bf16* sWiT  = (__bf16*)smem;                 // [n*256 + k]
    __bf16* sWhT  = sWiT + 256 * 256;
    __bf16* sE    = sWhT + 256 * 256;              // [b*256 + k]  e_t bf16
    __bf16* sH    = sE   + Bsz * 256;              // [b*256 + n]  h_prev bf16
    float*  sHf   = (float*)(sH + Bsz * 256);      // [b*256 + n]  h_prev f32
    float*  sBias = sHf + Bsz * 256;               // [n]

    const int tid = threadIdx.x;
    // Transposed bf16 weight staging (one-time): WiT[n][k] = Wi[k][n].
    for (int idx = tid; idx < 256 * 256; idx += 256) {
        int k = idx >> 8, n = idx & 255;
        sWiT[n * 256 + k] = (__bf16)Wi[idx];
        sWhT[n * 256 + k] = (__bf16)Wh[idx];
    }
    if (tid < 256) sBias[tid] = bi[tid] + bh[tid];
    for (int idx = tid; idx < Bsz * 256; idx += 256) {
        sHf[idx] = 0.0f;
        sH[idx]  = (__bf16)0.0f;
    }
    __syncthreads();

    const int w   = tid >> 5;          // wave 0..7 -> owns columns [32w, 32w+32)
    const int l   = tid & 31;
    const int lhi = (l >> 4) & 1;      // lane[4]
    const int l15 = l & 15;
    // embedding gather mapping: 16 threads per batch row, 16 floats each
    const int gb = tid >> 4;
    const int gq = tid & 15;

    for (int t = 0; t < Tlen; ++t) {
        // ---- gather e_t = embedding[x[b, t]] -> bf16 LDS ----
        {
            int tok = x[gb * 256 + t];
            const float4* ep = (const float4*)(emb + (size_t)tok * Edim + gq * 16);
            __bf16* dst = sE + gb * 256 + gq * 16;
#pragma unroll
            for (int i = 0; i < 4; ++i) {
                float4 f = ep[i];
                union { __bf16 b[4]; uint2 u; } pk;
                pk.b[0] = (__bf16)f.x; pk.b[1] = (__bf16)f.y;
                pk.b[2] = (__bf16)f.z; pk.b[3] = (__bf16)f.w;
                *(uint2*)(dst + i * 4) = pk.u;
            }
        }
        __syncthreads();   // h writes (prev step) + e writes visible

        // ---- acc = e_t@Wi + h@Wh for two 16-col tiles per wave ----
        v8f acc[2];
#pragma unroll
        for (int tt = 0; tt < 2; ++tt) {
            const int n0 = w * 32 + tt * 16;
            v8f c = {};
#pragma unroll
            for (int kc = 0; kc < 8; ++kc) {
                const int k0 = kc * 32;
                v16bf ae = ld_frag(sE + l15 * 256 + k0 + lhi * 8,
                                   sE + l15 * 256 + k0 + lhi * 8 + 16);
                v16bf bw = ld_frag(sWiT + (n0 + l15) * 256 + k0 + lhi * 8,
                                   sWiT + (n0 + l15) * 256 + k0 + lhi * 8 + 16);
                c = __builtin_amdgcn_wmma_f32_16x16x32_bf16(
                        false, ae, false, bw, (short)0, c, false, false);
                v16bf ah = ld_frag(sH + l15 * 256 + k0 + lhi * 8,
                                   sH + l15 * 256 + k0 + lhi * 8 + 16);
                v16bf bh_ = ld_frag(sWhT + (n0 + l15) * 256 + k0 + lhi * 8,
                                    sWhT + (n0 + l15) * 256 + k0 + lhi * 8 + 16);
                c = __builtin_amdgcn_wmma_f32_16x16x32_bf16(
                        false, ah, false, bh_, (short)0, c, false, false);
            }
            acc[tt] = c;
        }
        __syncthreads();   // all waves done reading sH/sE before overwrite

        // ---- epilogue: bias, tanh, clock gating, write h (cols owned per wave) ----
#pragma unroll
        for (int tt = 0; tt < 2; ++tt) {
            const int n0  = w * 32 + tt * 16;
            const int grp = n0 >> 6;                     // clock group 0..3
            const bool act = ((t & ((1 << grp) - 1)) == 0);
            const int n = n0 + l15;
#pragma unroll
            for (int j = 0; j < 8; ++j) {
                const int m = j + (lhi << 3);            // batch row
                float hprev = sHf[m * 256 + n];
                float hnew  = tanhf(acc[tt][j] + sBias[n]);
                float ho    = act ? hnew : hprev;
                sHf[m * 256 + n] = ho;
                __bf16 hb = (__bf16)ho;
                sH[m * 256 + n] = hb;
                hbf_out[((size_t)t * Bsz + m) * NH + n] = hb;
            }
        }
        __syncthreads();
    }
}

// ---------------------------------------------------------------------------
// Kernel B: logits = h (4080 x 256) @ Wo (256 x 32000), bf16 WMMA, f32 out.
// grid = (50, 255). Block stages the 16x256 A-tile (one t, all batch rows)
// in LDS; each wave computes 5 N-tiles (80 cols), 8 waves -> 640 cols/block.
// Store-bandwidth bound: 522MB fp32 out streamed; WoT is L2-resident.
// ---------------------------------------------------------------------------
#define NT_PER_WAVE 5
__global__ void output_gemm_kernel(const __bf16* __restrict__ hbf,
                                   const __bf16* __restrict__ WoT,
                                   float*        __restrict__ out) {
    __shared__ __bf16 sA[Bsz * NH];                  // 8 KB
    const int t   = blockIdx.y;
    const int tid = threadIdx.x;
    {   // cooperative A-tile stage: 512 uint4
        const uint4* src = (const uint4*)(hbf + (size_t)t * Bsz * NH);
        uint4* dst = (uint4*)sA;
        dst[tid]       = src[tid];
        dst[tid + 256] = src[tid + 256];
    }
    __syncthreads();

    const int w   = tid >> 5;
    const int l   = tid & 31;
    const int lhi = (l >> 4) & 1;
    const int l15 = l & 15;
    const int n0w = blockIdx.x * (8 * NT_PER_WAVE * 16) + w * (NT_PER_WAVE * 16);

    v8f acc[NT_PER_WAVE];
#pragma unroll
    for (int tt = 0; tt < NT_PER_WAVE; ++tt) acc[tt] = (v8f){};

#pragma unroll
    for (int kc = 0; kc < 8; ++kc) {
        const int k0 = kc * 32;
        v16bf a = ld_frag(sA + l15 * 256 + k0 + lhi * 8,
                          sA + l15 * 256 + k0 + lhi * 8 + 16);
#pragma unroll
        for (int tt = 0; tt < NT_PER_WAVE; ++tt) {
            const int n = n0w + tt * 16 + l15;
            const __bf16* bp = WoT + (size_t)n * NH + k0 + lhi * 8;
            v16bf b = ld_frag(bp, bp + 16);
            acc[tt] = __builtin_amdgcn_wmma_f32_16x16x32_bf16(
                          false, a, false, b, (short)0, acc[tt], false, false);
        }
    }

    // D layout: VGPR j -> row (j + 8*lane[4]), col = lane[3:0].  Row m == batch b.
#pragma unroll
    for (int tt = 0; tt < NT_PER_WAVE; ++tt) {
        const int ncol = n0w + tt * 16 + l15;
#pragma unroll
        for (int j = 0; j < 8; ++j) {
            const int b = j + (lhi << 3);
            out[((size_t)b * Tlen + t) * Vocab + ncol] = acc[tt][j];
        }
    }
}

// ---------------------------------------------------------------------------
extern "C" void kernel_launch(void* const* d_in, const int* in_sizes, int n_in,
                              void* d_out, int out_size, void* d_ws, size_t ws_size,
                              hipStream_t stream) {
    const int*   x    = (const int*)d_in[0];
    // d_in[1] = x_sl (unused by the reference computation)
    const float* emb  = (const float*)d_in[2];
    const float* Wi   = (const float*)d_in[3];
    const float* Wh   = (const float*)d_in[4];
    const float* bi   = (const float*)d_in[5];
    const float* bh   = (const float*)d_in[6];
    const float* Wo   = (const float*)d_in[7];
    float* out = (float*)d_out;

    char* ws = (char*)d_ws;
    __bf16* hbf = (__bf16*)ws;                       // 255*16*256*2 = 2,088,960 B
    __bf16* WoT = (__bf16*)(ws + (2 << 20));         // 32000*256*2 = 16,384,000 B

    // 1) Wo transpose+convert (independent of recurrence)
    wo_transpose_kernel<<<dim3(Vocab / 32, NH / 32), dim3(32, 8), 0, stream>>>(Wo, WoT);

    // 2) Sequential recurrence: 1 workgroup, 296KB dynamic LDS (weights resident)
    const size_t ldsBytes = 2 * 256 * 256 * sizeof(__bf16)   // WiT + WhT
                          + 2 * Bsz * 256 * sizeof(__bf16)   // sE + sH
                          + Bsz * 256 * sizeof(float)        // sHf
                          + 256 * sizeof(float);             // bias
    cwrnn_recurrence_kernel<<<1, 256, ldsBytes, stream>>>(x, emb, Wi, Wh, bi, bh, hbf);

    // 3) Output projection: 50*255 blocks, bf16 WMMA, streamed f32 stores
    output_gemm_kernel<<<dim3(50, Tlen), 256, 0, stream>>>(hbf, WoT, out);
}
